// URA_40200893890832
// MI455X (gfx1250) — matrix operand order repaired and checked by reading, not compiled, not run on hardware
//
#include <hip/hip_runtime.h>
#include <hip/hip_bf16.h>

typedef __attribute__((ext_vector_type(16))) _Float16 v16h;
typedef __attribute__((ext_vector_type(8)))  float    v8f;

#define BATCH 8
#define CH    64
#define HW    192
#define WS12  12
#define NPOS  144          // 12*12 tokens per window
#define NPIX  (BATCH*HW*HW)

__device__ __forceinline__ v8f wmma32(v16h a, v16h b, v8f c) {
  return __builtin_amdgcn_wmma_f32_16x16x32_f16(false, a, false, b, (short)0, c, false, false);
}

// ---- fragment helpers ----------------------------------------------------
// B matrices are staged "fragment-ready": fragment f holds 32 lanes x 16
// contiguous halves at base + f*512 + lane*16. Load = 2x ds_load_b128.
__device__ __forceinline__ v16h ldFragB(const _Float16* p) {
  return *(const v16h*)p;
}

// Destination offset (in halves) for element (k,n) of a fragment-ready B
// matrix with NT n-tiles: frag = (k>>5)*NT + (n>>4);
// lane = (n&15) + 16*((k>>3)&1); e = (k&7) + 8*((k>>4)&1).
__device__ __forceinline__ int bFragAddr(int k, int n, int NT) {
  return (((k >> 5) * NT + (n >> 4)) << 9) + (((n & 15) + (((k >> 3) & 1) << 4)) << 4) + (k & 7) + (((k >> 4) & 1) << 3);
}

// A fragment (16x32, f16), A[m][k] row-major with leading dim ld (halves).
// Lane layout (ISA 7.12.2): M = mBase + (lane&15); K = kBase + 8*(lane>>4) + (e&7) + 16*(e>>3)
// -> two contiguous 8-half runs per lane => 2x ds_load_b128.
__device__ __forceinline__ v16h ldsA(const _Float16* base, int ld, int mBase, int kBase,
                                     int lane, int kLim) {
  v16h a;
  const int m = mBase + (lane & 15);
  const int kOff = kBase + ((lane >> 4) << 3);
  #pragma unroll
  for (int e = 0; e < 16; ++e) {
    int k = kOff + (e & 7) + ((e >> 3) << 4);
    a[e] = (k < kLim) ? base[m * ld + k] : (_Float16)0.f;
  }
  return a;
}

// A fragment for per-window 3x3 conv: rows are token positions shifted by
// (dy,dx), zero outside the 12x12 window. av is [pos][ch], ld=64.
__device__ __forceinline__ v16h ldsAConvWin(const _Float16* av, int mBase, int kBase,
                                            int lane, int dy, int dx) {
  const int m = mBase + (lane & 15);
  const int py = m / WS12 + dy, px = m % WS12 + dx;
  const bool valid = ((unsigned)py < (unsigned)WS12) && ((unsigned)px < (unsigned)WS12);
  const int src = (py * WS12 + px) * CH;
  const int kOff = kBase + ((lane >> 4) << 3);
  v16h a;
  #pragma unroll
  for (int e = 0; e < 16; ++e) {
    int k = kOff + (e & 7) + ((e >> 3) << 4);
    a[e] = valid ? av[src + k] : (_Float16)0.f;
  }
  return a;
}

// A fragment for image conv: 16x16 output tile with 18-wide halo, ytile [pos18x18][ch].
__device__ __forceinline__ v16h ldsAConvImg(const _Float16* ytile, int mBase, int kBase,
                                            int lane, int dy, int dx) {
  const int m = mBase + (lane & 15);          // 0..255
  const int src = ((m / 16 + dy) * 18 + (m % 16 + dx)) * CH;
  const int kOff = kBase + ((lane >> 4) << 3);
  v16h a;
  #pragma unroll
  for (int e = 0; e < 16; ++e) {
    int k = kOff + (e & 7) + ((e >> 3) << 4);
    a[e] = ytile[src + k];
  }
  return a;
}

// ---------------- cg path -------------------------------------------------

__global__ void k_zero_max(unsigned* maxslot) { *maxslot = 0u; }

__global__ void k_cg0(const float* __restrict__ ms, float* __restrict__ cg0) {
  int idx = blockIdx.x * 256 + threadIdx.x;
  if (idx >= NPIX) return;
  int b = idx / (HW * HW), rr = idx % (HW * HW), oy = rr / HW, ox = rr % HW;
  float fy = (oy + 0.5f) * 0.25f - 0.5f;
  float fx = (ox + 0.5f) * 0.25f - 0.5f;
  int y0 = (int)floorf(fy), x0 = (int)floorf(fx);
  float ty = fy - (float)y0, tx = fx - (float)x0;
  int y0c = min(max(y0, 0), 47), y1c = min(max(y0 + 1, 0), 47);
  int x0c = min(max(x0, 0), 47), x1c = min(max(x0 + 1, 0), 47);
  const float* mb = ms + b * 48 * 48;
  float v00 = mb[y0c * 48 + x0c], v01 = mb[y0c * 48 + x1c];
  float v10 = mb[y1c * 48 + x0c], v11 = mb[y1c * 48 + x1c];
  float v = (1.f - ty) * ((1.f - tx) * v00 + tx * v01) + ty * ((1.f - tx) * v10 + tx * v11);
  float s = 1.f / (1.f + expf(-v));
  cg0[idx] = 0.5f - fabsf(s - 0.5f);
}

__global__ void k_blur(const float* __restrict__ cg0, float* __restrict__ cgraw,
                       unsigned* __restrict__ maxslot) {
  __shared__ float red[256];
  const float k1[7] = {0.004433048f, 0.054005582f, 0.242036228f, 0.399050284f,
                       0.242036228f, 0.054005582f, 0.004433048f};
  int idx = blockIdx.x * 256 + threadIdx.x;
  float val = 0.f;
  if (idx < NPIX) {
    int b = idx / (HW * HW), rr = idx % (HW * HW), oy = rr / HW, ox = rr % HW;
    const float* cb = cg0 + (size_t)b * HW * HW;
    float sum = 0.f;
    #pragma unroll
    for (int dy = -3; dy <= 3; ++dy) {
      int yy = oy + dy; yy = yy < 0 ? -yy : (yy > 191 ? 382 - yy : yy);
      float ky = k1[dy + 3];
      #pragma unroll
      for (int dx = -3; dx <= 3; ++dx) {
        int xx = ox + dx; xx = xx < 0 ? -xx : (xx > 191 ? 382 - xx : xx);
        sum += ky * k1[dx + 3] * cb[yy * HW + xx];
      }
    }
    val = sum * 4.f;
    cgraw[idx] = val;
  }
  red[threadIdx.x] = val;
  __syncthreads();
  for (int s = 128; s > 0; s >>= 1) {
    if (threadIdx.x < s) red[threadIdx.x] = fmaxf(red[threadIdx.x], red[threadIdx.x + s]);
    __syncthreads();
  }
  if (threadIdx.x == 0) atomicMax(maxslot, __float_as_uint(red[0]));  // values >= 0
}

__global__ void k_norm(const float* __restrict__ cgraw, const unsigned* __restrict__ maxslot,
                       float* __restrict__ cg) {
  int idx = blockIdx.x * 256 + threadIdx.x;
  if (idx >= NPIX) return;
  cg[idx] = cgraw[idx] / __uint_as_float(*maxslot);
}

// ---------------- per-window attention + conv1 + residual -----------------
// LDS arena (halves):
//  phase2: X[0,9216) L[9216,18432) Wt[18432,30720) -> Q[30720,39936)
//          Kfrag[39936,49152) Vfrag[49152,59392)   (V has 5 k-tiles, zero-padded)
//  phase3: attn f16 [m][n] ld144 at 0 (20736)
//  phase4: av [p][c] at 30720 (reuses Q)
//  phase5: w1frag [tap][frag] at 39936 (36864) -> arena = 76800 halves = 153600 B
#define OFF_X    0
#define OFF_L    9216
#define OFF_W    18432
#define OFF_Q    30720
#define OFF_K    39936
#define OFF_V    49152
#define OFF_ATTN 0
#define OFF_AV   30720
#define OFF_W1   39936

__global__ __launch_bounds__(256, 4) void k_window(
    const float* __restrict__ x, const float* __restrict__ l,
    const float* __restrict__ Wq, const float* __restrict__ bq,
    const float* __restrict__ Wk, const float* __restrict__ bk,
    const float* __restrict__ Wv, const float* __restrict__ bv,
    const float* __restrict__ w1, const float* __restrict__ b1,
    _Float16* __restrict__ y) {
  extern __shared__ char smem[];
  _Float16* sh = (_Float16*)smem;
  const int tid = threadIdx.x, lane = tid & 31;
  const int wave = __builtin_amdgcn_readfirstlane(tid >> 5);   // provably uniform
  const int wi = blockIdx.x;
  const int b = wi >> 8, wy = (wi >> 4) & 15, wx = wi & 15;

  // stage X, L windows as [pos][ch] f16
  for (int idx = tid; idx < NPOS * CH; idx += 256) {
    int c = idx / NPOS, p = idx % NPOS;
    int h = wy * WS12 + p / WS12, w = wx * WS12 + p % WS12;
    size_t g = (((size_t)b * CH + c) * HW + h) * HW + w;
    sh[OFF_X + p * CH + c] = (_Float16)x[g];
    sh[OFF_L + p * CH + c] = (_Float16)l[g];
  }
  // stage Wq/Wk/Wv as fragment-ready B:  B[k=j][n=c] = W[c][j]  (NT=4)
  for (int idx = tid; idx < CH * CH; idx += 256) {
    int c = idx >> 6, j = idx & 63;   // coalesced read of W[c*64+j]
    int dst = bFragAddr(j, c, 4);
    sh[OFF_W + 0 * 4096 + dst] = (_Float16)Wq[idx];
    sh[OFF_W + 1 * 4096 + dst] = (_Float16)Wk[idx];
    sh[OFF_W + 2 * 4096 + dst] = (_Float16)Wv[idx];
  }
  // zero the K=144..159 pad of V's fragment region (frags 16..19)
  for (int idx = tid; idx < 2048; idx += 256) sh[OFF_V + (16 << 9) + idx] = (_Float16)0.f;
  __syncthreads();

  // phase 2a: Q = X @ WqT + bq -> Q[p][c].  18 nt-pairs, shared A fragment.
  for (int pp = wave; pp < 18; pp += 8) {
    int mt = pp >> 1, nt0 = (pp & 1) << 1;
    v8f acc0 = {}, acc1 = {};
    #pragma unroll
    for (int ks = 0; ks < 2; ++ks) {
      v16h a = ldsA(sh + OFF_X, CH, mt * 16, ks * 32, lane, CH);
      const _Float16* bp = sh + OFF_W + ((ks * 4 + nt0) << 9) + (lane << 4);
      acc0 = wmma32(a, ldFragB(bp), acc0);
      acc1 = wmma32(a, ldFragB(bp + 512), acc1);
    }
    int n0 = nt0 * 16 + (lane & 15);
    float bias0 = bq[n0], bias1 = bq[n0 + 16];
    #pragma unroll
    for (int j = 0; j < 8; ++j) {
      int m = mt * 16 + j + ((lane >> 4) << 3);
      sh[OFF_Q + m * CH + n0]      = (_Float16)(acc0[j] + bias0);
      sh[OFF_Q + m * CH + n0 + 16] = (_Float16)(acc1[j] + bias1);
    }
  }
  // phase 2b: K = L @ WkT + bk -> fragment-ready B for attn: B[k=ch][n=pos] (NT=9)
  for (int pp = wave; pp < 18; pp += 8) {
    int mt = pp >> 1, nt0 = (pp & 1) << 1;
    v8f acc0 = {}, acc1 = {};
    #pragma unroll
    for (int ks = 0; ks < 2; ++ks) {
      v16h a = ldsA(sh + OFF_L, CH, mt * 16, ks * 32, lane, CH);
      const _Float16* bp = sh + OFF_W + 4096 + ((ks * 4 + nt0) << 9) + (lane << 4);
      acc0 = wmma32(a, ldFragB(bp), acc0);
      acc1 = wmma32(a, ldFragB(bp + 512), acc1);
    }
    int n0 = nt0 * 16 + (lane & 15);
    float bias0 = bk[n0], bias1 = bk[n0 + 16];
    #pragma unroll
    for (int j = 0; j < 8; ++j) {
      int m = mt * 16 + j + ((lane >> 4) << 3);        // pos
      sh[OFF_K + bFragAddr(n0,      m, 9)] = (_Float16)(acc0[j] + bias0);
      sh[OFF_K + bFragAddr(n0 + 16, m, 9)] = (_Float16)(acc1[j] + bias1);
    }
  }
  // phase 2c: V = L @ WvT + bv -> fragment-ready B for av: B[k=pos][n=ch] (NT=4)
  for (int pp = wave; pp < 18; pp += 8) {
    int mt = pp >> 1, nt0 = (pp & 1) << 1;
    v8f acc0 = {}, acc1 = {};
    #pragma unroll
    for (int ks = 0; ks < 2; ++ks) {
      v16h a = ldsA(sh + OFF_L, CH, mt * 16, ks * 32, lane, CH);
      const _Float16* bp = sh + OFF_W + 8192 + ((ks * 4 + nt0) << 9) + (lane << 4);
      acc0 = wmma32(a, ldFragB(bp), acc0);
      acc1 = wmma32(a, ldFragB(bp + 512), acc1);
    }
    int n0 = nt0 * 16 + (lane & 15);
    float bias0 = bv[n0], bias1 = bv[n0 + 16];
    #pragma unroll
    for (int j = 0; j < 8; ++j) {
      int m = mt * 16 + j + ((lane >> 4) << 3);        // pos
      sh[OFF_V + bFragAddr(m, n0,      4)] = (_Float16)(acc0[j] + bias0);
      sh[OFF_V + bFragAddr(m, n0 + 16, 4)] = (_Float16)(acc1[j] + bias1);
    }
  }
  __syncthreads();

  // phase 3: attn[q][k] = 0.125 * Q @ Kfrag, stored f16 [m][n] ld=144.
  // 9 mt rows x (4 nt-pairs + 1 single) = 45 work items.
  for (int pp = wave; pp < 45; pp += 8) {
    int mt = pp / 5, q = pp % 5;
    if (q < 4) {
      int nt0 = q * 2;
      v8f acc0 = {}, acc1 = {};
      #pragma unroll
      for (int ks = 0; ks < 2; ++ks) {
        v16h a = ldsA(sh + OFF_Q, CH, mt * 16, ks * 32, lane, CH);
        const _Float16* bp = sh + OFF_K + ((ks * 9 + nt0) << 9) + (lane << 4);
        acc0 = wmma32(a, ldFragB(bp), acc0);
        acc1 = wmma32(a, ldFragB(bp + 512), acc1);
      }
      int nn0 = nt0 * 16 + (lane & 15);
      #pragma unroll
      for (int j = 0; j < 8; ++j) {
        int mm = mt * 16 + j + ((lane >> 4) << 3);
        sh[OFF_ATTN + mm * NPOS + nn0]      = (_Float16)(acc0[j] * 0.125f);
        sh[OFF_ATTN + mm * NPOS + nn0 + 16] = (_Float16)(acc1[j] * 0.125f);
      }
    } else {
      v8f acc = {};
      #pragma unroll
      for (int ks = 0; ks < 2; ++ks)
        acc = wmma32(ldsA(sh + OFF_Q, CH, mt * 16, ks * 32, lane, CH),
                     ldFragB(sh + OFF_K + ((ks * 9 + 8) << 9) + (lane << 4)), acc);
      int nn = 128 + (lane & 15);
      #pragma unroll
      for (int j = 0; j < 8; ++j) {
        int mm = mt * 16 + j + ((lane >> 4) << 3);
        sh[OFF_ATTN + mm * NPOS + nn] = (_Float16)(acc[j] * 0.125f);
      }
    }
  }
  __syncthreads();

  // phase 4: av = attn @ Vfrag (K=144, 5 K-steps; A masked, B pre-zeroed)
  for (int pp = wave; pp < 18; pp += 8) {
    int mt = pp >> 1, nt0 = (pp & 1) << 1;
    v8f acc0 = {}, acc1 = {};
    #pragma unroll
    for (int ks = 0; ks < 5; ++ks) {
      v16h a = ldsA(sh + OFF_ATTN, NPOS, mt * 16, ks * 32, lane, NPOS);
      const _Float16* bp = sh + OFF_V + ((ks * 4 + nt0) << 9) + (lane << 4);
      acc0 = wmma32(a, ldFragB(bp), acc0);
      acc1 = wmma32(a, ldFragB(bp + 512), acc1);
    }
    int n0 = nt0 * 16 + (lane & 15);
    #pragma unroll
    for (int j = 0; j < 8; ++j) {
      int m = mt * 16 + j + ((lane >> 4) << 3);
      sh[OFF_AV + m * CH + n0]      = (_Float16)acc0[j];
      sh[OFF_AV + m * CH + n0 + 16] = (_Float16)acc1[j];
    }
  }
  __syncthreads();

  // stage w1 as fragment-ready B per tap: B[k=ic][n=oc] = w1[oc][ic][tap]
  for (int idx = tid; idx < CH * CH * 9; idx += 256) {
    int oc = idx / 576, icp = idx % 576, ic = icp / 9, tap = icp % 9;
    sh[OFF_W1 + tap * 4096 + bFragAddr(ic, oc, 4)] = (_Float16)w1[idx];
  }
  __syncthreads();

  // phase 5: per-window 3x3 conv (implicit GEMM, 9 taps) + bias + relu + residual
  for (int pp = wave; pp < 18; pp += 8) {
    int mt = pp >> 1, nt0 = (pp & 1) << 1;
    v8f acc0 = {}, acc1 = {};
    for (int tap = 0; tap < 9; ++tap) {
      int dy = tap / 3 - 1, dx = tap % 3 - 1;
      #pragma unroll
      for (int ks = 0; ks < 2; ++ks) {
        v16h a = ldsAConvWin(sh + OFF_AV, mt * 16, ks * 32, lane, dy, dx);
        const _Float16* bp = sh + OFF_W1 + tap * 4096 + ((ks * 4 + nt0) << 9) + (lane << 4);
        acc0 = wmma32(a, ldFragB(bp), acc0);
        acc1 = wmma32(a, ldFragB(bp + 512), acc1);
      }
    }
    int n0 = nt0 * 16 + (lane & 15), n1 = n0 + 16;
    float bias0 = b1[n0], bias1 = b1[n1];
    #pragma unroll
    for (int j = 0; j < 8; ++j) {
      int m = mt * 16 + j + ((lane >> 4) << 3);
      int h = wy * WS12 + m / WS12, w = wx * WS12 + m % WS12;
      size_t pixBase = ((size_t)b * HW + h) * HW + w;
      float r0 = acc0[j] + bias0; r0 = r0 > 0.f ? r0 : 0.f;
      float r1 = acc1[j] + bias1; r1 = r1 > 0.f ? r1 : 0.f;
      float xv0 = x[((size_t)b * CH + n0) * HW * HW + (size_t)h * HW + w];
      float xv1 = x[((size_t)b * CH + n1) * HW * HW + (size_t)h * HW + w];
      y[(pixBase << 6) + n0] = (_Float16)(xv0 + r0);   // NHWC f16
      y[(pixBase << 6) + n1] = (_Float16)(xv1 + r1);
    }
  }
}

// ---------------- image conv3 (+relu) + 1x1 conv4 -------------------------
// LDS: ytile 18x18x64 f16 [0,20736), w3frag [20736,57600) halves, then 256 f32 outacc.
__global__ __launch_bounds__(256, 4) void k_conv34(
    const _Float16* __restrict__ y, const float* __restrict__ w3,
    const float* __restrict__ b3, const float* __restrict__ w4,
    const float* __restrict__ b4, float* __restrict__ xo, float* __restrict__ outp) {
  extern __shared__ char smem[];
  _Float16* sh = (_Float16*)smem;
  float* outacc = (float*)(smem + 57600 * 2);
  const int tid = threadIdx.x, lane = tid & 31;
  const int wave = __builtin_amdgcn_readfirstlane(tid >> 5);   // provably uniform
  const int bi = blockIdx.x, b = bi / 144, r = bi % 144, ty = r / 12, tx = r % 12;
  const int gy0 = ty * 16 - 1, gx0 = tx * 16 - 1;

  for (int idx = tid; idx < 324 * CH; idx += 256) {
    int pos = idx >> 6, c = idx & 63;
    int gy = gy0 + pos / 18, gx = gx0 + pos % 18;
    _Float16 v = (_Float16)0.f;
    if ((unsigned)gy < (unsigned)HW && (unsigned)gx < (unsigned)HW)
      v = y[((((size_t)b * HW + gy) * HW + gx) << 6) + c];
    sh[idx] = v;
  }
  for (int idx = tid; idx < CH * CH * 9; idx += 256) {
    int oc = idx / 576, icp = idx % 576, ic = icp / 9, tap = icp % 9;
    sh[20736 + tap * 4096 + bFragAddr(ic, oc, 4)] = (_Float16)w3[idx];
  }
  if (tid < 256) outacc[tid] = 0.f;
  __syncthreads();

  // 16 mt x 4 nt = 64 tiles -> 32 nt-pairs with shared A fragment
  for (int pp = wave; pp < 32; pp += 8) {
    int mt = pp >> 1, nt0 = (pp & 1) << 1;
    v8f acc0 = {}, acc1 = {};
    for (int tap = 0; tap < 9; ++tap) {
      int dy = tap / 3, dx = tap % 3;
      #pragma unroll
      for (int ks = 0; ks < 2; ++ks) {
        v16h a = ldsAConvImg(sh, mt * 16, ks * 32, lane, dy, dx);
        const _Float16* bp = sh + 20736 + tap * 4096 + ((ks * 4 + nt0) << 9) + (lane << 4);
        acc0 = wmma32(a, ldFragB(bp), acc0);
        acc1 = wmma32(a, ldFragB(bp + 512), acc1);
      }
    }
    int n0 = nt0 * 16 + (lane & 15), n1 = n0 + 16;
    float bias0 = b3[n0], bias1 = b3[n1];
    float w4c0 = w4[n0], w4c1 = w4[n1];
    #pragma unroll
    for (int j = 0; j < 8; ++j) {
      int m = mt * 16 + j + ((lane >> 4) << 3);
      int gy = ty * 16 + m / 16, gx = tx * 16 + m % 16;
      float v0 = acc0[j] + bias0; v0 = v0 > 0.f ? v0 : 0.f;
      float v1 = acc1[j] + bias1; v1 = v1 > 0.f ? v1 : 0.f;
      xo[(((size_t)b * CH + n0) * HW + gy) * HW + gx] = v0;
      xo[(((size_t)b * CH + n1) * HW + gy) * HW + gx] = v1;
      atomicAdd(&outacc[m], v0 * w4c0 + v1 * w4c1);  // ds_add_f32
    }
  }
  __syncthreads();
  if (tid < 256) {
    int gy = ty * 16 + tid / 16, gx = tx * 16 + tid % 16;
    outp[((size_t)b * HW + gy) * HW + gx] = outacc[tid] + b4[0];
  }
}

// ---------------- launcher ------------------------------------------------
extern "C" void kernel_launch(void* const* d_in, const int* in_sizes, int n_in,
                              void* d_out, int out_size, void* d_ws, size_t ws_size,
                              hipStream_t stream) {
  const float* x  = (const float*)d_in[0];
  const float* l  = (const float*)d_in[1];
  const float* ms = (const float*)d_in[2];
  const float* Wq = (const float*)d_in[3];
  const float* bq = (const float*)d_in[4];
  const float* Wk = (const float*)d_in[5];
  const float* bk = (const float*)d_in[6];
  const float* Wv = (const float*)d_in[7];
  const float* bv = (const float*)d_in[8];
  const float* w1 = (const float*)d_in[9];
  const float* b1 = (const float*)d_in[10];
  const float* w3 = (const float*)d_in[11];
  const float* b3 = (const float*)d_in[12];
  const float* w4 = (const float*)d_in[13];
  const float* b4 = (const float*)d_in[14];

  float* xo   = (float*)d_out;                       // (8,64,192,192)
  float* outp = xo + (size_t)BATCH * CH * HW * HW;   // (8,1,192,192)
  float* cg   = outp + (size_t)NPIX;                 // (8,1,192,192)

  char* ws = (char*)d_ws;
  _Float16* y    = (_Float16*)ws;                              // NHWC f16, 37,748,736 B
  float* cg0     = (float*)(ws + 37748736);                    // 1,179,648 B
  float* cgraw   = (float*)(ws + 37748736 + 1179648);          // 1,179,648 B
  unsigned* mslt = (unsigned*)(ws + 37748736 + 2 * 1179648);   // 4 B

  const int pxBlocks = (NPIX + 255) / 256;
  k_zero_max<<<1, 1, 0, stream>>>(mslt);
  k_cg0 <<<pxBlocks, 256, 0, stream>>>(ms, cg0);
  k_blur<<<pxBlocks, 256, 0, stream>>>(cg0, cgraw, mslt);
  k_norm<<<pxBlocks, 256, 0, stream>>>(cgraw, mslt, cg);

  k_window<<<2048, 256, 76800 * sizeof(_Float16), stream>>>(
      x, l, Wq, bq, Wk, bk, Wv, bv, w1, b1, y);

  k_conv34<<<BATCH * 12 * 12, 256, 57600 * 2 + 256 * 4, stream>>>(
      y, w3, b3, w4, b4, xo, outp);
}
